// Transformer_RGBD_69947837382775
// MI455X (gfx1250) — matrix operand order repaired and checked by reading, not compiled, not run on hardware
//
#include <hip/hip_runtime.h>
#include <hip/hip_bf16.h>
#include <cstdint>

// ---------------------------------------------------------------------------
// Dual-stream (RGB-D) transformer encoder layer for MI455X (gfx1250).
// - All matmuls: v_wmma_f32_16x16x32_bf16 (bf16 in, f32 accumulate)
// - GEMM K-tiles: double-buffered GLOBAL_LOAD_ASYNC_TO_LDS_B128 (ASYNCcnt)
// - Attention K-tiles: Tensor Data Mover (tensor_load_to_lds) with LDS pad
// - Flash-style online softmax; with ALPHA=BETA=0.5:
//     o_img = 0.5*(A_img@V + A_dpt@V),  o_dpt = 0.5*(A_dpt@Vd + A_img@Vd)
// ---------------------------------------------------------------------------

#define S_LEN 1024
#define BB    8
#define DD    512
#define HH    8
#define HDIM  64
#define FFD   2048
#define MROWS (BB * S_LEN)   // 8192

typedef __attribute__((ext_vector_type(16))) __bf16 v16bf;
typedef __attribute__((ext_vector_type(8)))  float  v8f;
typedef __attribute__((ext_vector_type(4)))  unsigned int u32x4;
typedef __attribute__((ext_vector_type(4)))  int          i32x4;
typedef __attribute__((ext_vector_type(8)))  int          i32x8;

union FragU { unsigned int u[8]; v16bf v; };

__device__ __forceinline__ unsigned short f2bf(float x) {
    union { float f; unsigned int u; } c; c.f = x;
    unsigned int r = c.u + 0x7FFFu + ((c.u >> 16) & 1u);
    return (unsigned short)(r >> 16);
}
__device__ __forceinline__ unsigned int pack2(float a, float b) {
    return (unsigned int)f2bf(a) | ((unsigned int)f2bf(b) << 16);
}

// ---- CDNA5 async global->LDS copy (per-lane 16B), tracked by ASYNCcnt -----
__device__ __forceinline__ void async_b128(unsigned lds_off, unsigned long long ga) {
    asm volatile("global_load_async_to_lds_b128 %0, %1, off"
                 :: "v"(lds_off), "v"(ga) : "memory");
}
template <int N>
__device__ __forceinline__ void wait_async() {
    asm volatile("s_wait_asynccnt %0" :: "i"(N) : "memory");
}

// ---- CDNA5 Tensor Data Mover: 2D tile (32 rows x 64 bf16, stride 64) into
// LDS with padding: 16B pad every 128B -> LDS row stride 72 elements. --------
__device__ __forceinline__ void tdm_load_k_tile(unsigned lds_addr,
                                                unsigned long long gaddr) {
    u32x4 g0;
    g0[0] = 1u;                                           // count=1 (valid D#)
    g0[1] = lds_addr;                                     // lds_addr [63:32]
    g0[2] = (unsigned)(gaddr & 0xFFFFFFFFu);              // global_addr lo
    g0[3] = (unsigned)((gaddr >> 32) & 0x01FFFFFFu)       // global_addr [56:32]
          | (2u << 30);                                   // type=2 (image)
    i32x8 g1;
    g1[0] = (1 << 16)      // data_size = 2 bytes
          | (1 << 20)      // pad_enable
          | (4 << 22)      // pad_interval: 32 DWORDs (=128B) before pad
          | (3 << 25);     // pad_amount:   4 DWORDs (=16B)
    g1[1] = (int)(64u << 16);   // tensor_dim0[15:0]=64 in bits [63:48]
    g1[2] = (int)(32u << 16);   // tensor_dim0 hi=0 | tensor_dim1[15:0]=32
    g1[3] = (int)(64u << 16);   // tensor_dim1 hi=0 | tile_dim0=64
    g1[4] = 32;                 // tile_dim1=32 | tile_dim2=0
    g1[5] = 64;                 // tensor_dim0_stride = 64 elements
    g1[6] = 0;
    g1[7] = 0;
    i32x4 z4 = (i32x4){0, 0, 0, 0};
#if defined(__clang_major__) && (__clang_major__ >= 23)
    i32x8 z8 = (i32x8){0, 0, 0, 0, 0, 0, 0, 0};
    __builtin_amdgcn_tensor_load_to_lds(g0, g1, z4, z4, z8, 0);
#else
    __builtin_amdgcn_tensor_load_to_lds(g0, g1, z4, z4, 0);
#endif
}

// ------------------------------- prep kernels ------------------------------

__global__ void prep_act_kernel(const float* __restrict__ src,
                                const float* __restrict__ srcd,
                                const float* __restrict__ pos,
                                unsigned short* __restrict__ x_bf,
                                unsigned short* __restrict__ xd_bf,
                                unsigned short* __restrict__ qk_bf) {
    int g = blockIdx.x * blockDim.x + threadIdx.x;
    int i = g * 4;                       // flat index in [B,S,D]
    int r = i >> 9;
    int d = i & 511;
    int b = r >> 10;
    int s = r & 1023;
    size_t j = ((size_t)(s * BB + b) << 9) + d;   // [S,B,D] flat
    float4 xs  = *(const float4*)(src  + j);
    float4 xds = *(const float4*)(srcd + j);
    float4 ps  = *(const float4*)(pos  + j);
    uint2 o;
    o.x = pack2(xs.x, xs.y);  o.y = pack2(xs.z, xs.w);
    *(uint2*)(x_bf + i) = o;
    o.x = pack2(xds.x, xds.y); o.y = pack2(xds.z, xds.w);
    *(uint2*)(xd_bf + i) = o;
    o.x = pack2(xs.x + ps.x, xs.y + ps.y); o.y = pack2(xs.z + ps.z, xs.w + ps.w);
    *(uint2*)(qk_bf + i) = o;
}

__global__ void cvt_bf16_kernel(const float* __restrict__ in,
                                unsigned short* __restrict__ out) {
    int i = (blockIdx.x * blockDim.x + threadIdx.x) * 4;
    float4 v = *(const float4*)(in + i);
    uint2 o;
    o.x = pack2(v.x, v.y); o.y = pack2(v.z, v.w);
    *(uint2*)(out + i) = o;
}

// ------------------------------- GEMM kernel -------------------------------
// Y[M,N] = act( alpha*(X[M,K] @ W[N,K]^T + bias) ) + resid
// Block tile 128x128, 8 waves, wave tile 32x64 (2x4 WMMA tiles), K step 32.
// Double-buffered async-to-LDS staging.
// flags: 1=relu, 2=resid plain [M,N], 4=resid in [S,B,D] layout (N==512),
//        8=bf16 out scattered to heads [B,H,S,HD] (N==512)

#define GBM 128
#define GBN 128
#define GLP 40   // padded LDS K-stride (ushort elements)

__global__ __launch_bounds__(256) void gemm_bf16_kernel(
    const unsigned short* __restrict__ A,   // [M,K] bf16
    const unsigned short* __restrict__ Bw,  // [N,K] bf16 (torch Linear weight)
    const float* __restrict__ bias,         // [N] or null
    float* __restrict__ outF,               // [M,N] f32 or null
    unsigned short* __restrict__ outB,      // bf16 out or null
    const float* __restrict__ resid,        // residual or null
    int M, int N, int K, float alpha, int flags) {

    __shared__ unsigned short lA[2][GBM * GLP];
    __shared__ unsigned short lB[2][GBN * GLP];

    const int tid  = threadIdx.x;
    const int m0   = blockIdx.y * GBM;
    const int n0   = blockIdx.x * GBN;
    const int wave = tid >> 5;
    const int lane = tid & 31;
    const int lm   = lane & 15;
    const int hi   = lane >> 4;
    const int wm   = wave >> 1;      // 0..3 -> M offset wm*32
    const int wn   = wave & 1;       // 0..1 -> N offset wn*64

    int ak[8], bk[8];
#pragma unroll
    for (int j = 0; j < 8; ++j) {
        ak[j] = 2 * j + 8 * hi + ((j >= 4) ? 8 : 0);
        bk[j] = 2 * j + 8 * hi + ((j >= 4) ? 8 : 0);
    }

    v8f acc[2][4];
#pragma unroll
    for (int mt = 0; mt < 2; ++mt)
#pragma unroll
        for (int nt = 0; nt < 4; ++nt) acc[mt][nt] = (v8f){};

    const int ldRow = tid >> 1;          // 0..127
    const int ldSeg = tid & 1;           // 0..1 (32B each)

    const unsigned ldsA0 = (unsigned)(uintptr_t)(&lA[0][ldRow * GLP + ldSeg * 16]);
    const unsigned ldsB0 = (unsigned)(uintptr_t)(&lB[0][ldRow * GLP + ldSeg * 16]);
    const unsigned long long gA0 =
        (unsigned long long)(uintptr_t)(A  + (size_t)(m0 + ldRow) * K + ldSeg * 16);
    const unsigned long long gB0 =
        (unsigned long long)(uintptr_t)(Bw + (size_t)(n0 + ldRow) * K + ldSeg * 16);

    auto issue_tile = [&](int k0, int buf) {
        unsigned la = ldsA0 + (unsigned)buf * (GBM * GLP * 2);
        unsigned lb = ldsB0 + (unsigned)buf * (GBN * GLP * 2);
        unsigned long long ga = gA0 + (unsigned long long)k0 * 2ull;
        unsigned long long gb = gB0 + (unsigned long long)k0 * 2ull;
        async_b128(la,      ga);
        async_b128(la + 32, ga + 32);
        async_b128(lb,      gb);
        async_b128(lb + 32, gb + 32);
    };

    const int nk = K >> 5;
    issue_tile(0, 0);

    for (int i = 0; i < nk; ++i) {
        if (i + 1 < nk) {
            issue_tile((i + 1) << 5, (i + 1) & 1);
            wait_async<4>();          // tile i has landed
        } else {
            wait_async<0>();
        }
        __syncthreads();

        const unsigned short* Abase = lA[i & 1];
        const unsigned short* Bbase = lB[i & 1];

        FragU af[2], bf[4];
#pragma unroll
        for (int mt = 0; mt < 2; ++mt) {
            const unsigned short* base = Abase + (wm * 32 + mt * 16 + lm) * GLP;
#pragma unroll
            for (int j = 0; j < 8; ++j)
                af[mt].u[j] = *(const unsigned int*)(base + ak[j]);
        }
#pragma unroll
        for (int nt = 0; nt < 4; ++nt) {
            const unsigned short* base = Bbase + (wn * 64 + nt * 16 + lm) * GLP;
#pragma unroll
            for (int j = 0; j < 8; ++j)
                bf[nt].u[j] = *(const unsigned int*)(base + bk[j]);
        }
#pragma unroll
        for (int mt = 0; mt < 2; ++mt)
#pragma unroll
            for (int nt = 0; nt < 4; ++nt)
                acc[mt][nt] = __builtin_amdgcn_wmma_f32_16x16x32_bf16(
                    false, af[mt].v, false, bf[nt].v, (short)0, acc[mt][nt],
                    false, false);
        __syncthreads();
    }

    // epilogue
#pragma unroll
    for (int mt = 0; mt < 2; ++mt) {
#pragma unroll
        for (int nt = 0; nt < 4; ++nt) {
#pragma unroll
            for (int j = 0; j < 8; ++j) {
                int row = m0 + wm * 32 + mt * 16 + j + 8 * hi;
                int col = n0 + wn * 64 + nt * 16 + lm;
                float v = acc[mt][nt][j];
                if (bias) v += bias[col];
                v *= alpha;
                if (flags & 1) v = fmaxf(v, 0.0f);
                if (flags & 2) v += resid[(size_t)row * N + col];
                if (flags & 4) {
                    int b = row >> 10, s = row & 1023;
                    v += resid[((size_t)(s * BB + b) << 9) + col];
                }
                if (outF) outF[(size_t)row * N + col] = v;
                if (outB) {
                    if (flags & 8) {
                        int h = col >> 6, hd = col & 63;
                        int b = row >> 10, s = row & 1023;
                        outB[(((size_t)(b * HH + h) << 10) + s) * HDIM + hd] = f2bf(v);
                    } else {
                        outB[(size_t)row * N + col] = f2bf(v);
                    }
                }
            }
        }
    }
}

// ---------------------------- flash attention ------------------------------
// Q,K,V1,V2: [B*H, S, HD] bf16 (Q pre-scaled by 1/sqrt(HD)).
// O1 = softmax(Q K^T) @ V1,  O2 = softmax(Q K^T) @ V2   (f32, same layout)
// Block: 256 threads = 8 waves, each wave owns 16 query rows (128 rows/blk).
// K chunk staged by the Tensor Data Mover (padded), V chunks transposed by ALU.

#define KLP 72   // lK stride: [s'][hd] — produced by TDM pad (128B + 16B)
#define VLP 40   // lV stride: [hd][s']
#define PLP 40   // lP stride: [m][k]

__global__ __launch_bounds__(256) void attn_kernel(
    const unsigned short* __restrict__ Q,
    const unsigned short* __restrict__ Kt,
    const unsigned short* __restrict__ V1,
    const unsigned short* __restrict__ V2,
    float* __restrict__ O1,
    float* __restrict__ O2) {

    __shared__ unsigned short lK[32 * KLP];
    __shared__ unsigned short lV1[64 * VLP];
    __shared__ unsigned short lV2[64 * VLP];
    __shared__ unsigned short lP[8][16 * PLP];

    const int tid  = threadIdx.x;
    const int bh   = blockIdx.x >> 3;            // which (b,h)
    const int rb   = (blockIdx.x & 7) * 128;     // row-block base
    const int wave = tid >> 5;
    const int lane = tid & 31;
    const int lm   = lane & 15;
    const int hi   = lane >> 4;
    const int row0 = rb + wave * 16;

    int ak[8], bk[8];
#pragma unroll
    for (int j = 0; j < 8; ++j) {
        ak[j] = 2 * j + 8 * hi + ((j >= 4) ? 8 : 0);
        bk[j] = 2 * j + 8 * hi + ((j >= 4) ? 8 : 0);
    }

    // preload Q fragments (rows row0..row0+15, full HD=64 -> 2 k-steps)
    FragU qf[2];
    {
        const unsigned short* qbase = Q + (((size_t)bh << 10) + row0 + lm) * HDIM;
#pragma unroll
        for (int ks = 0; ks < 2; ++ks)
#pragma unroll
            for (int j = 0; j < 8; ++j)
                qf[ks].u[j] = *(const unsigned int*)(qbase + ks * 32 + ak[j]);
    }

    v8f o1a[4], o2a[4];
#pragma unroll
    for (int t = 0; t < 4; ++t) { o1a[t] = (v8f){}; o2a[t] = (v8f){}; }
    float m_run[8], l_run[8];
#pragma unroll
    for (int j = 0; j < 8; ++j) { m_run[j] = -1e30f; l_run[j] = 0.0f; }

    const unsigned lK_base = (unsigned)(uintptr_t)(&lK[0]);
    const int vRow = tid & 31, vSeg = tid >> 5;  // V staging: transpose

    for (int s0 = 0; s0 < S_LEN; s0 += 32) {
        // ---- TDM: DMA K chunk (32 x 64 bf16) into lK with stride-72 padding
        if (tid == 0) {
            tdm_load_k_tile(lK_base,
                (unsigned long long)(uintptr_t)(Kt + (((size_t)bh << 10) + s0) * HDIM));
        }
        // ---- stage V1/V2 transposed ([hd][s']) with plain loads
        {
            union { uint4 q; unsigned short h[8]; } u;
            u.q = *(const uint4*)(V1 + (((size_t)bh << 10) + s0 + vRow) * HDIM + vSeg * 8);
#pragma unroll
            for (int i = 0; i < 8; ++i) lV1[(vSeg * 8 + i) * VLP + vRow] = u.h[i];
            u.q = *(const uint4*)(V2 + (((size_t)bh << 10) + s0 + vRow) * HDIM + vSeg * 8);
#pragma unroll
            for (int i = 0; i < 8; ++i) lV2[(vSeg * 8 + i) * VLP + vRow] = u.h[i];
        }
        if (tid == 0) __builtin_amdgcn_s_wait_tensorcnt(0);
        __syncthreads();

        // ---- scores: 16 x 32 (two 16x16 tiles), K-dim = HD = 2 x 32
        v8f sc[2]; sc[0] = (v8f){}; sc[1] = (v8f){};
#pragma unroll
        for (int ks = 0; ks < 2; ++ks) {
#pragma unroll
            for (int nt = 0; nt < 2; ++nt) {
                FragU kb;
                const unsigned short* base = lK + (nt * 16 + lm) * KLP + ks * 32;
#pragma unroll
                for (int j = 0; j < 8; ++j)
                    kb.u[j] = *(const unsigned int*)(base + bk[j]);
                sc[nt] = __builtin_amdgcn_wmma_f32_16x16x32_bf16(
                    false, qf[ks].v, false, kb.v, (short)0, sc[nt], false, false);
            }
        }

        // ---- online softmax per C-row (row = j + 8*hi)
#pragma unroll
        for (int j = 0; j < 8; ++j) {
            float mx = fmaxf(sc[0][j], sc[1][j]);
#pragma unroll
            for (int off = 1; off < 16; off <<= 1)
                mx = fmaxf(mx, __shfl_xor(mx, off));
            float mnew = fmaxf(m_run[j], mx);
            float corr = __expf(m_run[j] - mnew);
            m_run[j] = mnew;
            float p0 = __expf(sc[0][j] - mnew);
            float p1 = __expf(sc[1][j] - mnew);
            float rs = p0 + p1;
#pragma unroll
            for (int off = 1; off < 16; off <<= 1)
                rs += __shfl_xor(rs, off);
            l_run[j] = l_run[j] * corr + rs;
#pragma unroll
            for (int t = 0; t < 4; ++t) { o1a[t][j] *= corr; o2a[t][j] *= corr; }
            lP[wave][(j + 8 * hi) * PLP + 0 * 16 + lm] = f2bf(p0);
            lP[wave][(j + 8 * hi) * PLP + 1 * 16 + lm] = f2bf(p1);
        }
        __syncthreads();

        // ---- P @ V chunk: A = P (16x32), B = V chunk (32 x 64) -> 4 N tiles
        FragU pf;
        {
            const unsigned short* base = lP[wave] + lm * PLP;
#pragma unroll
            for (int j = 0; j < 8; ++j)
                pf.u[j] = *(const unsigned int*)(base + ak[j]);
        }
#pragma unroll
        for (int nt = 0; nt < 4; ++nt) {
            FragU vb;
            const unsigned short* b1 = lV1 + (nt * 16 + lm) * VLP;
#pragma unroll
            for (int j = 0; j < 8; ++j)
                vb.u[j] = *(const unsigned int*)(b1 + bk[j]);
            o1a[nt] = __builtin_amdgcn_wmma_f32_16x16x32_bf16(
                false, pf.v, false, vb.v, (short)0, o1a[nt], false, false);
            const unsigned short* b2 = lV2 + (nt * 16 + lm) * VLP;
#pragma unroll
            for (int j = 0; j < 8; ++j)
                vb.u[j] = *(const unsigned int*)(b2 + bk[j]);
            o2a[nt] = __builtin_amdgcn_wmma_f32_16x16x32_bf16(
                false, pf.v, false, vb.v, (short)0, o2a[nt], false, false);
        }
        __syncthreads();
    }

    // ---- normalize and store
#pragma unroll
    for (int j = 0; j < 8; ++j) {
        float inv = 1.0f / l_run[j];
        int row = row0 + j + 8 * hi;
#pragma unroll
        for (int nt = 0; nt < 4; ++nt) {
            int col = nt * 16 + lm;
            size_t idx = (((size_t)bh << 10) + row) * HDIM + col;
            O1[idx] = o1a[nt][j] * inv;
            O2[idx] = o2a[nt][j] * inv;
        }
    }
}

// --------------------------- combine attention heads -----------------------
__global__ void combine_heads_kernel(const float* __restrict__ T1,
                                     const float* __restrict__ T2,
                                     unsigned short* __restrict__ outB,
                                     float w) {
    int g = blockIdx.x * blockDim.x + threadIdx.x;
    int i = g * 4;
    int r = i >> 9;
    int e = i & 511;
    int h = e >> 6, hd = e & 63;
    int b = r >> 10, s = r & 1023;
    size_t x = (((size_t)(b * HH + h) << 10) + s) * HDIM + hd;
    float4 a = *(const float4*)(T1 + x);
    float4 c = *(const float4*)(T2 + x);
    float wa = 1.0f - w;
    uint2 o;
    o.x = pack2(wa * a.x + w * c.x, wa * a.y + w * c.y);
    o.y = pack2(wa * a.z + w * c.z, wa * a.w + w * c.w);
    *(uint2*)(outB + i) = o;
}

// -------------------------------- LayerNorm --------------------------------
__global__ __launch_bounds__(128) void ln_kernel(
    const float* __restrict__ X,
    const float* __restrict__ gamma,
    const float* __restrict__ beta,
    float* __restrict__ outF,
    unsigned short* __restrict__ outB,
    float* __restrict__ outT) {

    __shared__ float sS[4], sQ[4];
    int row = blockIdx.x;
    int tid = threadIdx.x;
    int lane = tid & 31, wave = tid >> 5;
    int c = tid * 4;
    float4 v = *(const float4*)(X + (size_t)row * DD + c);
    float s = v.x + v.y + v.z + v.w;
    float q = v.x * v.x + v.y * v.y + v.z * v.z + v.w * v.w;
#pragma unroll
    for (int off = 1; off < 32; off <<= 1) {
        s += __shfl_xor(s, off);
        q += __shfl_xor(q, off);
    }
    if (lane == 0) { sS[wave] = s; sQ[wave] = q; }
    __syncthreads();
    s = sS[0] + sS[1] + sS[2] + sS[3];
    q = sQ[0] + sQ[1] + sQ[2] + sQ[3];
    float mean = s * (1.0f / DD);
    float var  = q * (1.0f / DD) - mean * mean;
    float inv  = rsqrtf(var + 1e-5f);
    float4 g4 = *(const float4*)(gamma + c);
    float4 b4 = *(const float4*)(beta + c);
    float4 y;
    y.x = (v.x - mean) * inv * g4.x + b4.x;
    y.y = (v.y - mean) * inv * g4.y + b4.y;
    y.z = (v.z - mean) * inv * g4.z + b4.z;
    y.w = (v.w - mean) * inv * g4.w + b4.w;
    if (outF) *(float4*)(outF + (size_t)row * DD + c) = y;
    if (outB) {
        uint2 o; o.x = pack2(y.x, y.y); o.y = pack2(y.z, y.w);
        *(uint2*)(outB + (size_t)row * DD + c) = o;
    }
    if (outT) {
        int b = row >> 10, sI = row & 1023;
        *(float4*)(outT + ((size_t)(sI * BB + b) << 9) + c) = y;
    }
}

// ------------------------------- host driver -------------------------------

extern "C" void kernel_launch(void* const* d_in, const int* in_sizes, int n_in,
                              void* d_out, int out_size, void* d_ws, size_t ws_size,
                              hipStream_t stream) {
    const float* src       = (const float*)d_in[0];
    const float* srcd      = (const float*)d_in[1];
    const float* pos       = (const float*)d_in[2];
    const float* aw_img    = (const float*)d_in[3];
    const float* ab_img    = (const float*)d_in[4];
    const float* aw_dpt    = (const float*)d_in[5];
    const float* ab_dpt    = (const float*)d_in[6];
    const float* w1i       = (const float*)d_in[7];
    const float* b1i       = (const float*)d_in[8];
    const float* w2i       = (const float*)d_in[9];
    const float* b2i       = (const float*)d_in[10];
    const float* w1d       = (const float*)d_in[11];
    const float* b1d       = (const float*)d_in[12];
    const float* w2d       = (const float*)d_in[13];
    const float* b2d       = (const float*)d_in[14];
    const float* ln_img    = (const float*)d_in[15];
    const float* ln_dpt    = (const float*)d_in[16];
    float* out = (float*)d_out;

    const size_t MD  = (size_t)MROWS * DD;     // 4,194,304
    const size_t WAE = (size_t)4 * DD * DD;    // 1,048,576
    const size_t WFE = (size_t)FFD * DD;       // 1,048,576
    const size_t MF  = (size_t)MROWS * FFD;    // 16,777,216

    char* p = (char*)d_ws;
    auto alloc = [&](size_t bytes) -> char* {
        char* r = p; p += (bytes + 255) & ~(size_t)255; return r;
    };

    unsigned short* qk_bf  = (unsigned short*)alloc(MD * 2);
    unsigned short* x_bf   = (unsigned short*)alloc(MD * 2);
    unsigned short* xd_bf  = (unsigned short*)alloc(MD * 2);
    unsigned short* wai    = (unsigned short*)alloc(WAE * 2);
    unsigned short* wad    = (unsigned short*)alloc(WAE * 2);
    unsigned short* w1i_b  = (unsigned short*)alloc(WFE * 2);
    unsigned short* w2i_b  = (unsigned short*)alloc(WFE * 2);
    unsigned short* w1d_b  = (unsigned short*)alloc(WFE * 2);
    unsigned short* w2d_b  = (unsigned short*)alloc(WFE * 2);
    unsigned short* q_bf   = (unsigned short*)alloc(MD * 2);
    unsigned short* k_bf   = (unsigned short*)alloc(MD * 2);
    unsigned short* v_bf   = (unsigned short*)alloc(MD * 2);
    unsigned short* qd_bf  = (unsigned short*)alloc(MD * 2);
    unsigned short* kd_bf  = (unsigned short*)alloc(MD * 2);
    unsigned short* vd_bf  = (unsigned short*)alloc(MD * 2);
    float* t1 = (float*)alloc(MD * 4);   // A_img @ V
    float* t2 = (float*)alloc(MD * 4);   // A_dpt @ V
    float* t3 = (float*)alloc(MD * 4);   // A_dpt @ Vd
    float* t4 = (float*)alloc(MD * 4);   // A_img @ Vd
    unsigned short* oimg_bf = (unsigned short*)alloc(MD * 2);
    unsigned short* odpt_bf = (unsigned short*)alloc(MD * 2);
    float* res1i = (float*)alloc(MD * 4);
    float* res1d = (float*)alloc(MD * 4);
    float* ln1i_f = (float*)alloc(MD * 4);
    float* ln1d_f = (float*)alloc(MD * 4);
    unsigned short* ln1i_b = (unsigned short*)alloc(MD * 2);
    unsigned short* ln1d_b = (unsigned short*)alloc(MD * 2);
    unsigned short* h_i = (unsigned short*)alloc(MF * 2);
    unsigned short* h_d = (unsigned short*)alloc(MF * 2);
    float* res2i = (float*)alloc(MD * 4);
    float* res2d = (float*)alloc(MD * 4);

    // ---- stage 0: activations + weight conversion
    prep_act_kernel<<<MD / 1024, 256, 0, stream>>>(src, srcd, pos, x_bf, xd_bf, qk_bf);
    cvt_bf16_kernel<<<WAE / 1024, 256, 0, stream>>>(aw_img, wai);
    cvt_bf16_kernel<<<WAE / 1024, 256, 0, stream>>>(aw_dpt, wad);
    cvt_bf16_kernel<<<WFE / 1024, 256, 0, stream>>>(w1i, w1i_b);
    cvt_bf16_kernel<<<WFE / 1024, 256, 0, stream>>>(w2i, w2i_b);
    cvt_bf16_kernel<<<WFE / 1024, 256, 0, stream>>>(w1d, w1d_b);
    cvt_bf16_kernel<<<WFE / 1024, 256, 0, stream>>>(w2d, w2d_b);

    const float qscale = 0.125f;  // 1/sqrt(HD)
    dim3 gProj(DD / GBN, MROWS / GBM);        // (4, 64)
    dim3 gFfn1(FFD / GBN, MROWS / GBM);       // (16, 64)

    // ---- stage 1: QKV projections (bf16 heads layout out)
    gemm_bf16_kernel<<<gProj, 256, 0, stream>>>(qk_bf, wai + 0 * DD * DD, ab_img + 0 * DD,
        nullptr, q_bf, nullptr, MROWS, DD, DD, qscale, 8);
    gemm_bf16_kernel<<<gProj, 256, 0, stream>>>(qk_bf, wai + 1 * DD * DD, ab_img + 1 * DD,
        nullptr, k_bf, nullptr, MROWS, DD, DD, 1.0f, 8);
    gemm_bf16_kernel<<<gProj, 256, 0, stream>>>(x_bf,  wai + 2 * DD * DD, ab_img + 2 * DD,
        nullptr, v_bf, nullptr, MROWS, DD, DD, 1.0f, 8);
    gemm_bf16_kernel<<<gProj, 256, 0, stream>>>(xd_bf, wad + 0 * DD * DD, ab_dpt + 0 * DD,
        nullptr, qd_bf, nullptr, MROWS, DD, DD, qscale, 8);
    gemm_bf16_kernel<<<gProj, 256, 0, stream>>>(xd_bf, wad + 1 * DD * DD, ab_dpt + 1 * DD,
        nullptr, kd_bf, nullptr, MROWS, DD, DD, 1.0f, 8);
    gemm_bf16_kernel<<<gProj, 256, 0, stream>>>(xd_bf, wad + 2 * DD * DD, ab_dpt + 2 * DD,
        nullptr, vd_bf, nullptr, MROWS, DD, DD, 1.0f, 8);

    // ---- stage 2: flash attention (each call produces P@V and P@Vd)
    attn_kernel<<<BB * HH * (S_LEN / 128), 256, 0, stream>>>(q_bf,  k_bf,  v_bf, vd_bf, t1, t4);
    attn_kernel<<<BB * HH * (S_LEN / 128), 256, 0, stream>>>(qd_bf, kd_bf, v_bf, vd_bf, t2, t3);
    // o_img = (1-ALPHA)*t1 + ALPHA*t2 ; o_dpt = (1-BETA)*t3 + BETA*t4
    combine_heads_kernel<<<MD / 1024, 256, 0, stream>>>(t1, t2, oimg_bf, 0.5f);
    combine_heads_kernel<<<MD / 1024, 256, 0, stream>>>(t3, t4, odpt_bf, 0.5f);

    // ---- stage 3: output projections + residual (src in [S,B,D] layout)
    gemm_bf16_kernel<<<gProj, 256, 0, stream>>>(oimg_bf, wai + 3 * DD * DD, ab_img + 3 * DD,
        res1i, nullptr, src,  MROWS, DD, DD, 1.0f, 4);
    gemm_bf16_kernel<<<gProj, 256, 0, stream>>>(odpt_bf, wad + 3 * DD * DD, ab_dpt + 3 * DD,
        res1d, nullptr, srcd, MROWS, DD, DD, 1.0f, 4);

    // ---- stage 4: LN1
    ln_kernel<<<MROWS, 128, 0, stream>>>(res1i, ln_img + 0 * DD, ln_img + 1 * DD,
                                         ln1i_f, ln1i_b, nullptr);
    ln_kernel<<<MROWS, 128, 0, stream>>>(res1d, ln_dpt + 0 * DD, ln_dpt + 1 * DD,
                                         ln1d_f, ln1d_b, nullptr);

    // ---- stage 5: FFN up + ReLU
    gemm_bf16_kernel<<<gFfn1, 256, 0, stream>>>(ln1i_b, w1i_b, b1i,
        nullptr, h_i, nullptr, MROWS, FFD, DD, 1.0f, 1);
    gemm_bf16_kernel<<<gFfn1, 256, 0, stream>>>(ln1d_b, w1d_b, b1d,
        nullptr, h_d, nullptr, MROWS, FFD, DD, 1.0f, 1);

    // ---- stage 6: FFN down + residual (plain layout)
    gemm_bf16_kernel<<<gProj, 256, 0, stream>>>(h_i, w2i_b, b2i,
        res2i, nullptr, ln1i_f, MROWS, DD, FFD, 1.0f, 2);
    gemm_bf16_kernel<<<gProj, 256, 0, stream>>>(h_d, w2d_b, b2d,
        res2d, nullptr, ln1d_f, MROWS, DD, FFD, 1.0f, 2);

    // ---- stage 7: LN2 -> d_out (transposed back to [S,B,D]); img then dpt
    ln_kernel<<<MROWS, 128, 0, stream>>>(res2i, ln_img + 2 * DD, ln_img + 3 * DD,
                                         nullptr, nullptr, out);
    ln_kernel<<<MROWS, 128, 0, stream>>>(res2d, ln_dpt + 2 * DD, ln_dpt + 3 * DD,
                                         nullptr, nullptr, out + MD);
}
